// SpatialAttentionLayer_54047868452939
// MI455X (gfx1250) — matrix operand order repaired and checked by reading, not compiled, not run on hardware
//
#include <hip/hip_runtime.h>
#include <hip/hip_bf16.h>

typedef __attribute__((ext_vector_type(16))) __bf16 v16bf;
typedef __attribute__((ext_vector_type(8)))  __bf16 v8bf;
typedef __attribute__((ext_vector_type(8)))  float  v8f;

#define N_PIX 4096
#define C_IN  256
#define CQ    32
#define B_SZ  4

// ---------------------------------------------------------------------------
// Kernel 1: transpose + convert  xT[b][n][c] = bf16(x[b][c][n])
// ---------------------------------------------------------------------------
__global__ void xpose_cvt(const float* __restrict__ x, unsigned short* __restrict__ xT_) {
  __shared__ float t[32][33];
  __bf16* xT = reinterpret_cast<__bf16*>(xT_);
  int b  = blockIdx.z;
  int c0 = blockIdx.y * 32;
  int n0 = blockIdx.x * 32;
  int tx = threadIdx.x;     // 0..31
  int ty = threadIdx.y;     // 0..7
  const float* xb = x + (size_t)b * C_IN * N_PIX;
  #pragma unroll
  for (int i = 0; i < 4; ++i)
    t[ty + 8*i][tx] = xb[(size_t)(c0 + ty + 8*i) * N_PIX + n0 + tx];
  __syncthreads();
  __bf16* xTb = xT + (size_t)b * N_PIX * C_IN;
  #pragma unroll
  for (int i = 0; i < 4; ++i)
    xTb[(size_t)(n0 + ty + 8*i) * C_IN + c0 + tx] = (__bf16)t[tx][ty + 8*i];
}

// ---------------------------------------------------------------------------
// Kernel 2: pack weights into one bf16 [320 x 256] matrix + f32 bias[320]
// rows 0..31 = wq, 32..63 = wk, 64..319 = wv
// ---------------------------------------------------------------------------
__global__ void wcvt(const float* __restrict__ wq, const float* __restrict__ wk,
                     const float* __restrict__ wv, const float* __restrict__ bq,
                     const float* __restrict__ bk, const float* __restrict__ bv,
                     unsigned short* __restrict__ wh_, float* __restrict__ bfull) {
  __bf16* wh = reinterpret_cast<__bf16*>(wh_);
  int r = blockIdx.x;    // 0..319
  int t = threadIdx.x;   // 0..255
  const float* src; float bias;
  if (r < 32)      { src = wq + (size_t)r * 256;        bias = bq[r];      }
  else if (r < 64) { src = wk + (size_t)(r-32) * 256;   bias = bk[r-32];   }
  else             { src = wv + (size_t)(r-64) * 256;   bias = bv[r-64];   }
  wh[(size_t)r * 256 + t] = (__bf16)src[t];
  if (t == 0) bfull[r] = bias;
}

// ---------------------------------------------------------------------------
// Kernel 3: projections via WMMA.  D[o,n] = sum_c W[o,c] * x[c,n] + bias[o]
// q,k stored [n][32] (attention A-layout source), v stored [c][n].
// grid (64 ntiles, 20 mtiles, B), block 128 (4 waves; wave w owns n-subtile w)
// ---------------------------------------------------------------------------
__global__ void proj(const unsigned short* __restrict__ xT_,
                     const unsigned short* __restrict__ wh_,
                     const float* __restrict__ bfull,
                     unsigned short* __restrict__ qh_, unsigned short* __restrict__ kh_,
                     unsigned short* __restrict__ vh_) {
  const __bf16* xT = reinterpret_cast<const __bf16*>(xT_);
  const __bf16* wh = reinterpret_cast<const __bf16*>(wh_);
  __bf16* qh = reinterpret_cast<__bf16*>(qh_);
  __bf16* kh = reinterpret_cast<__bf16*>(kh_);
  __bf16* vh = reinterpret_cast<__bf16*>(vh_);

  int b     = blockIdx.z;
  int mbase = blockIdx.y * 16;
  int lane  = threadIdx.x & 31;
  int w     = threadIdx.x >> 5;
  int nbase = blockIdx.x * 64 + w * 16;
  int hi    = lane >> 4;
  int ln    = lane & 15;

  const __bf16* xTb = xT + (size_t)b * N_PIX * C_IN;

  v8f acc = {};
  for (int cb = 0; cb < C_IN; cb += 32) {
    // A operand: weight tile 16(M=o) x 32(K=c). Lane = row M; two 16B chunks.
    union { v16bf v; v8bf h[2]; } A;
    const __bf16* ap = wh + (size_t)(mbase + ln) * C_IN + cb + 8*hi;
    A.h[0] = *(const v8bf*)(ap);
    A.h[1] = *(const v8bf*)(ap + 16);
    // B operand: xT tile 32(K=c) x 16(N=n). Lane = col n; two 16B chunks.
    union { v16bf v; v8bf h[2]; } Bu;
    const __bf16* bp = xTb + (size_t)(nbase + ln) * C_IN + cb + 8*hi;
    Bu.h[0] = *(const v8bf*)(bp);
    Bu.h[1] = *(const v8bf*)(bp + 16);
    v16bf Bm = Bu.v;
    acc = __builtin_amdgcn_wmma_f32_16x16x32_bf16(false, A.v, false, Bm,
                                                  (short)0, acc, false, false);
  }
  // D layout: VGPR r holds M = r + 8*hi, N = ln
  #pragma unroll
  for (int r = 0; r < 8; ++r) {
    int orow = mbase + r + 8*hi;
    float val = acc[r] + bfull[orow];
    int ncol = nbase + ln;
    if (orow < 32)      qh[((size_t)b * N_PIX + ncol) * CQ + orow]            = (__bf16)val;
    else if (orow < 64) kh[((size_t)b * N_PIX + ncol) * CQ + (orow - 32)]     = (__bf16)val;
    else                vh[((size_t)b * C_IN + (orow - 64)) * N_PIX + ncol]   = (__bf16)val;
  }
}

// ---------------------------------------------------------------------------
// Kernel 4: fused attention. One workgroup = (batch, 32-row i-tile) so every
// K-tile load feeds 2 score WMMAs and every V-tile load feeds 2 PV WMMAs.
// 4 waves: wave w owns j-subtile w of each 64-wide j-strip (scores) and
// output channels [64w, 64w+64) (P·V). Two-pass softmax (max, then exp/sum+PV).
// ---------------------------------------------------------------------------
__global__ void attn(const unsigned short* __restrict__ qh_,
                     const unsigned short* __restrict__ kh_,
                     const unsigned short* __restrict__ vh_,
                     const float* __restrict__ x, float* __restrict__ out) {
  __shared__ __bf16 sP[32][64];     // P strip (2 i-subtiles) in A-friendly layout
  __shared__ float  sRed[4][32];
  __shared__ float  sM[32];
  __shared__ float  sL[32];

  const __bf16* qh = reinterpret_cast<const __bf16*>(qh_);
  const __bf16* kh = reinterpret_cast<const __bf16*>(kh_);
  const __bf16* vh = reinterpret_cast<const __bf16*>(vh_);

  int b     = blockIdx.x >> 7;      // grid = B * 128
  int itile = blockIdx.x & 127;     // 32-row tile index
  int lane  = threadIdx.x & 31;
  int w     = threadIdx.x >> 5;
  int hi    = lane >> 4;
  int ln    = lane & 15;

  const __bf16* qb = qh + (size_t)b * N_PIX * CQ;
  const __bf16* kb = kh + (size_t)b * N_PIX * CQ;
  const __bf16* vb = vh + (size_t)b * C_IN * N_PIX;

  // Q A-tiles 16(i) x 32(c), two i-subtiles: persistent in registers
  union { v16bf v; v8bf h[2]; } AQ[2];
  #pragma unroll
  for (int ii = 0; ii < 2; ++ii) {
    const __bf16* qp = qb + (size_t)(itile*32 + ii*16 + ln) * CQ + 8*hi;
    AQ[ii].h[0] = *(const v8bf*)(qp);
    AQ[ii].h[1] = *(const v8bf*)(qp + 16);
  }

  // ---------------- pass 1: row max ----------------
  float rmax[2][8];
  #pragma unroll
  for (int ii = 0; ii < 2; ++ii)
    #pragma unroll
    for (int r = 0; r < 8; ++r) rmax[ii][r] = -3.0e38f;

  for (int js = 0; js < 64; ++js) {
    int j0 = js*64 + w*16;
    const __bf16* kp = kb + (size_t)(j0 + ln) * CQ + 8*hi;   // B: 32(K=c) x 16(N=j)
    union { v16bf v; v8bf h[2]; } BKu;
    BKu.h[0] = *(const v8bf*)(kp);
    BKu.h[1] = *(const v8bf*)(kp + 16);
    v16bf BK = BKu.v;
    #pragma unroll
    for (int ii = 0; ii < 2; ++ii) {
      v8f S = {};
      S = __builtin_amdgcn_wmma_f32_16x16x32_bf16(false, AQ[ii].v, false, BK,
                                                  (short)0, S, false, false);
      #pragma unroll
      for (int r = 0; r < 8; ++r) rmax[ii][r] = fmaxf(rmax[ii][r], S[r]);
    }
  }
  #pragma unroll
  for (int msk = 1; msk < 16; msk <<= 1)
    #pragma unroll
    for (int ii = 0; ii < 2; ++ii)
      #pragma unroll
      for (int r = 0; r < 8; ++r)
        rmax[ii][r] = fmaxf(rmax[ii][r], __shfl_xor(rmax[ii][r], msk, 32));
  if (ln == 0) {
    #pragma unroll
    for (int ii = 0; ii < 2; ++ii)
      #pragma unroll
      for (int r = 0; r < 8; ++r) sRed[w][ii*16 + r + 8*hi] = rmax[ii][r];
  }
  __syncthreads();
  if (threadIdx.x < 32)
    sM[threadIdx.x] = fmaxf(fmaxf(sRed[0][threadIdx.x], sRed[1][threadIdx.x]),
                            fmaxf(sRed[2][threadIdx.x], sRed[3][threadIdx.x]));
  __syncthreads();
  float mrow[2][8];
  #pragma unroll
  for (int ii = 0; ii < 2; ++ii)
    #pragma unroll
    for (int r = 0; r < 8; ++r) mrow[ii][r] = sM[ii*16 + r + 8*hi];

  // ---------------- pass 2: exp, row sum, P·V ----------------
  v8f acc[2][4] = {{{},{},{},{}}, {{},{},{},{}}};
  float lsum[2][8];
  #pragma unroll
  for (int ii = 0; ii < 2; ++ii)
    #pragma unroll
    for (int r = 0; r < 8; ++r) lsum[ii][r] = 0.f;

  for (int js = 0; js < 64; ++js) {
    int j0 = js*64 + w*16;
    const __bf16* kp = kb + (size_t)(j0 + ln) * CQ + 8*hi;
    union { v16bf v; v8bf h[2]; } BKu;
    BKu.h[0] = *(const v8bf*)(kp);
    BKu.h[1] = *(const v8bf*)(kp + 16);
    v16bf BK = BKu.v;
    #pragma unroll
    for (int ii = 0; ii < 2; ++ii) {
      v8f S = {};
      S = __builtin_amdgcn_wmma_f32_16x16x32_bf16(false, AQ[ii].v, false, BK,
                                                  (short)0, S, false, false);
      #pragma unroll
      for (int r = 0; r < 8; ++r) {
        float p = __expf(S[r] - mrow[ii][r]);
        lsum[ii][r] += p;
        sP[ii*16 + r + 8*hi][w*16 + ln] = (__bf16)p;   // D-layout -> strip
      }
    }
    __syncthreads();
    #pragma unroll
    for (int kk = 0; kk < 2; ++kk) {
      union { v16bf v; v8bf h[2]; } AP[2];     // A: P tiles 16(i) x 32(K=j)
      #pragma unroll
      for (int ii = 0; ii < 2; ++ii) {
        const __bf16* pp = &sP[ii*16 + ln][kk*32 + 8*hi];
        AP[ii].h[0] = *(const v8bf*)(pp);
        AP[ii].h[1] = *(const v8bf*)(pp + 16);
      }
      #pragma unroll
      for (int t = 0; t < 4; ++t) {
        int c = w*64 + t*16 + ln;
        const __bf16* vp = vb + (size_t)c * N_PIX + js*64 + kk*32 + 8*hi;
        union { v16bf v; v8bf h[2]; } BVu;
        BVu.h[0] = *(const v8bf*)(vp);
        BVu.h[1] = *(const v8bf*)(vp + 16);
        v16bf BV = BVu.v;        // B: 32(K=j) x 16(N=c), reused 2x
        acc[0][t] = __builtin_amdgcn_wmma_f32_16x16x32_bf16(false, AP[0].v, false, BV,
                                                            (short)0, acc[0][t], false, false);
        acc[1][t] = __builtin_amdgcn_wmma_f32_16x16x32_bf16(false, AP[1].v, false, BV,
                                                            (short)0, acc[1][t], false, false);
      }
    }
    __syncthreads();
  }

  // reduce row sums
  #pragma unroll
  for (int msk = 1; msk < 16; msk <<= 1)
    #pragma unroll
    for (int ii = 0; ii < 2; ++ii)
      #pragma unroll
      for (int r = 0; r < 8; ++r) lsum[ii][r] += __shfl_xor(lsum[ii][r], msk, 32);
  if (ln == 0) {
    #pragma unroll
    for (int ii = 0; ii < 2; ++ii)
      #pragma unroll
      for (int r = 0; r < 8; ++r) sRed[w][ii*16 + r + 8*hi] = lsum[ii][r];
  }
  __syncthreads();
  if (threadIdx.x < 32)
    sL[threadIdx.x] = sRed[0][threadIdx.x] + sRed[1][threadIdx.x] +
                      sRed[2][threadIdx.x] + sRed[3][threadIdx.x];
  __syncthreads();
  float rinv[2][8];
  #pragma unroll
  for (int ii = 0; ii < 2; ++ii)
    #pragma unroll
    for (int r = 0; r < 8; ++r) rinv[ii][r] = 1.0f / sL[ii*16 + r + 8*hi];

  // epilogue: divide by row sum, add residual; stream with non-temporal hints
  const float* xb = x   + (size_t)b * C_IN * N_PIX;
  float*       ob = out + (size_t)b * C_IN * N_PIX;
  #pragma unroll
  for (int ii = 0; ii < 2; ++ii)
    #pragma unroll
    for (int t = 0; t < 4; ++t) {
      int c = w*64 + t*16 + ln;
      #pragma unroll
      for (int r = 0; r < 8; ++r) {
        int i = itile*32 + ii*16 + r + 8*hi;
        size_t idx = (size_t)c * N_PIX + i;
        float res = __builtin_nontemporal_load(&xb[idx]);
        __builtin_nontemporal_store(acc[ii][t][r] * rinv[ii][r] + res, &ob[idx]);
      }
    }
}

// ---------------------------------------------------------------------------
extern "C" void kernel_launch(void* const* d_in, const int* in_sizes, int n_in,
                              void* d_out, int out_size, void* d_ws, size_t ws_size,
                              hipStream_t stream) {
  const float* x  = (const float*)d_in[0];
  const float* wq = (const float*)d_in[1];
  const float* bq = (const float*)d_in[2];
  const float* wk = (const float*)d_in[3];
  const float* bk = (const float*)d_in[4];
  const float* wv = (const float*)d_in[5];
  const float* bv = (const float*)d_in[6];
  float* out = (float*)d_out;

  // workspace carve (~18.3 MB total)
  char* p = (char*)d_ws;
  auto carve = [&](size_t bytes) { char* r = p; p += (bytes + 255) & ~(size_t)255; return r; };
  unsigned short* xT    = (unsigned short*)carve((size_t)B_SZ * N_PIX * C_IN * 2);
  unsigned short* wh    = (unsigned short*)carve((size_t)320 * 256 * 2);
  float*          bfull = (float*)carve((size_t)320 * 4);
  unsigned short* qh    = (unsigned short*)carve((size_t)B_SZ * N_PIX * CQ * 2);
  unsigned short* kh    = (unsigned short*)carve((size_t)B_SZ * N_PIX * CQ * 2);
  unsigned short* vh    = (unsigned short*)carve((size_t)B_SZ * C_IN * N_PIX * 2);

  xpose_cvt<<<dim3(N_PIX/32, C_IN/32, B_SZ), dim3(32, 8), 0, stream>>>(x, xT);
  wcvt<<<dim3(320), dim3(256), 0, stream>>>(wq, wk, wv, bq, bk, bv, wh, bfull);
  proj<<<dim3(N_PIX/64, 320/16, B_SZ), dim3(128), 0, stream>>>(xT, wh, bfull, qh, kh, vh);
  attn<<<dim3(B_SZ * 128), dim3(128), 0, stream>>>(qh, kh, vh, x, out);
}
